// VarGRU_17867063951796
// MI455X (gfx1250) — matrix operand order repaired and checked by reading, not compiled, not run on hardware
//
#include <hip/hip_runtime.h>

// ---------------------------------------------------------------------------
// VarGRU on gfx1250 (MI455X): bf16 WMMA recurrence, f32 accumulate.
//   T=512, B=64, E=512, H=1024, G=3H=3072.
//   Phase 1: gx = X@W_ih^T + b_ih (one big WMMA GEMM).
//   Phase 2: ONE persistent kernel (64 blocks, all co-resident) runs all T
//            recurrent steps with a device-wide atomic barrier between steps.
// ---------------------------------------------------------------------------

#define GRU_B 64
#define GRU_E 512
#define GRU_H 1024
#define GRU_G 3072   // 3*H
#define STEP_BLOCKS 64

typedef __attribute__((ext_vector_type(16))) __bf16    bf16x16;
typedef __attribute__((ext_vector_type(8)))  float     f32x8;
typedef __attribute__((ext_vector_type(4)))  float     f32x4;
typedef __attribute__((ext_vector_type(8)))  unsigned  u32x8;
typedef __attribute__((ext_vector_type(4)))  unsigned  u32x4;
typedef __attribute__((ext_vector_type(2)))  unsigned  u32x2;

// round-to-nearest-even f32 -> bf16
static __device__ __forceinline__ unsigned short f2bf(float f) {
    unsigned u = __builtin_bit_cast(unsigned, f);
    unsigned r = u + 0x7FFFu + ((u >> 16) & 1u);
    return (unsigned short)(r >> 16);
}

// Load a 16-element bf16 fragment as two 16B chunks (works for LDS or global).
static __device__ __forceinline__ bf16x16 load_frag(const unsigned short* p0,
                                                    const unsigned short* p1) {
    u32x4 lo = *(const u32x4*)p0;
    u32x4 hi = *(const u32x4*)p1;
    u32x8 w  = __builtin_shufflevector(lo, hi, 0, 1, 2, 3, 4, 5, 6, 7);
    return __builtin_bit_cast(bf16x16, w);
}

static __device__ __forceinline__ float fsigmoid(float x) {
    return 1.0f / (1.0f + __expf(-x));
}

// ---------------------------------------------------------------------------
// f32 -> bf16 bulk conversion (vectorized by 4; all sizes are multiples of 4)
// ---------------------------------------------------------------------------
__global__ void cvt_f32_to_bf16(const float* __restrict__ src,
                                unsigned short* __restrict__ dst, long n4) {
    long i      = (long)blockIdx.x * blockDim.x + threadIdx.x;
    long stride = (long)gridDim.x * blockDim.x;
    for (; i < n4; i += stride) {
        f32x4 f = ((const f32x4*)src)[i];
        u32x2 p;
        p.x = (unsigned)f2bf(f.x) | ((unsigned)f2bf(f.y) << 16);
        p.y = (unsigned)f2bf(f.z) | ((unsigned)f2bf(f.w) << 16);
        ((u32x2*)dst)[i] = p;
    }
}

// ---------------------------------------------------------------------------
// Phase 1:  gx[r][g] = sum_e xbf[r][e] * wih[g][e] + b_ih[g]
// One 16x16 output tile per wave, K = E = 512 (16 WMMA k-steps of 32).
// grid = rowTiles*24 blocks of 256 threads (8 waves).
// ---------------------------------------------------------------------------
__global__ __launch_bounds__(256) void gx_gemm(
        const unsigned short* __restrict__ xbf,
        const unsigned short* __restrict__ wih,
        const float* __restrict__ b_ih,
        float* __restrict__ gx) {
    const int wid  = threadIdx.x >> 5;
    const int lane = threadIdx.x & 31;
    const int l15  = lane & 15;
    const int kh   = lane >> 4;

    long tileId  = (long)blockIdx.x * 8 + wid;
    int  colTile = (int)(tileId % 192);          // G/16
    long rowTile = tileId / 192;

    const unsigned short* aBase = xbf + (rowTile * 16 + l15) * (long)GRU_E;
    const unsigned short* bBase = wih + ((long)colTile * 16 + l15) * (long)GRU_E;

    f32x8 acc = {0, 0, 0, 0, 0, 0, 0, 0};
#pragma unroll 4
    for (int k0 = 0; k0 < GRU_E; k0 += 32) {
        bf16x16 a = load_frag(aBase + k0 + 16 * kh, aBase + k0 + 16 * kh + 8);
        bf16x16 b = load_frag(bBase + k0 + 16 * kh, bBase + k0 + 16 * kh + 8);
        acc = __builtin_amdgcn_wmma_f32_16x16x32_bf16(
            false, a, false, b, (short)0, acc, false, false);
    }

    const int   col = colTile * 16 + l15;
    const float bi  = b_ih[col];
#pragma unroll
    for (int i = 0; i < 8; ++i) {
        long row = rowTile * 16 + i + 8 * kh;     // C layout: VGPR i -> M=i (+8 for hi half-wave)
        gx[row * (long)GRU_G + col] = acc[i] + bi;
    }
}

// ---------------------------------------------------------------------------
// Phase 2: persistent recurrence kernel.
// Grid: 64 blocks x 128 threads (4 waves, 32 KB LDS) -> all blocks resident,
// so a device-wide atomic barrier between steps cannot deadlock.
//   blockIdx.x = rowTile(0..3) + 4*colGroup(0..15); wave w -> colTile = cg*4+w
//   Each wave: one 16x16 column tile for all 3 gates (r/z/n share A fragment).
// ---------------------------------------------------------------------------
__global__ __launch_bounds__(128) void gru_persistent(
        const unsigned short* __restrict__ whh,   // [G][H] bf16
        const float* __restrict__ hx,             // [B][H] f32
        const float* __restrict__ gx,             // [T][B][G] f32 (x-gates + b_ih)
        const float* __restrict__ b_hh,           // [G]    f32
        float* __restrict__ out,                  // [T][B][H] f32
        int T,
        unsigned* __restrict__ cnt) {             // zero-initialized barrier counter
    __shared__ __align__(16) unsigned short sh[16 * GRU_H];   // 32 KB bf16 h-tile

    const int tid  = threadIdx.x;
    const int wid  = tid >> 5;
    const int lane = tid & 31;
    const int l15  = lane & 15;
    const int kh   = lane >> 4;

    const int rowTile = blockIdx.x & 3;
    const int colTile = (blockIdx.x >> 2) * 4 + wid;
    const int rowBase = rowTile * 16;
    const int colBase = colTile * 16;

    // Per-wave constants held in registers across all T steps.
    const unsigned short* aP  = sh + l15 * GRU_H;
    const unsigned short* bRp = whh + ((long)(0 * GRU_H + colBase + l15)) * GRU_H;
    const unsigned short* bZp = whh + ((long)(1 * GRU_H + colBase + l15)) * GRU_H;
    const unsigned short* bNp = whh + ((long)(2 * GRU_H + colBase + l15)) * GRU_H;

    const int   col = colBase + l15;
    const float bhr = b_hh[col];
    const float bhz = b_hh[GRU_H + col];
    const float bhn = b_hh[2 * GRU_H + col];

    for (int t = 0; t < T; ++t) {
        const float* hprev = (t == 0) ? hx : out + (size_t)(t - 1) * (GRU_B * GRU_H);
        const float* gx_t  = gx  + (size_t)t * (GRU_B * GRU_G);
        float*       out_t = out + (size_t)t * (GRU_B * GRU_H);

        // ---- stage h rows [rowBase, rowBase+16) as bf16 into LDS ----
        {
            const f32x4* src = (const f32x4*)(hprev + (long)rowBase * GRU_H);
            u32x2*       dst = (u32x2*)sh;
            for (int i = tid; i < (16 * GRU_H) / 4; i += 128) {
                f32x4 f = src[i];
                u32x2 p;
                p.x = (unsigned)f2bf(f.x) | ((unsigned)f2bf(f.y) << 16);
                p.y = (unsigned)f2bf(f.z) | ((unsigned)f2bf(f.w) << 16);
                dst[i] = p;
            }
        }
        __syncthreads();

        // ---- K-loop: 3 gates share one A fragment per k-step ----
        f32x8 accR = {0, 0, 0, 0, 0, 0, 0, 0};
        f32x8 accZ = {0, 0, 0, 0, 0, 0, 0, 0};
        f32x8 accN = {0, 0, 0, 0, 0, 0, 0, 0};

#pragma unroll 2
        for (int k0 = 0; k0 < GRU_H; k0 += 32) {
            bf16x16 a = load_frag(aP + k0 + 16 * kh, aP + k0 + 16 * kh + 8);

            // pull the next W_hh chunks toward L0 while we compute
            __builtin_prefetch(bRp + k0 + 128, 0, 1);
            __builtin_prefetch(bZp + k0 + 128, 0, 1);
            __builtin_prefetch(bNp + k0 + 128, 0, 1);

            bf16x16 b = load_frag(bRp + k0 + 16 * kh, bRp + k0 + 16 * kh + 8);
            accR = __builtin_amdgcn_wmma_f32_16x16x32_bf16(
                false, a, false, b, (short)0, accR, false, false);

            b = load_frag(bZp + k0 + 16 * kh, bZp + k0 + 16 * kh + 8);
            accZ = __builtin_amdgcn_wmma_f32_16x16x32_bf16(
                false, a, false, b, (short)0, accZ, false, false);

            b = load_frag(bNp + k0 + 16 * kh, bNp + k0 + 16 * kh + 8);
            accN = __builtin_amdgcn_wmma_f32_16x16x32_bf16(
                false, a, false, b, (short)0, accN, false, false);
        }

        // ---- fused gate epilogue ----
#pragma unroll
        for (int i = 0; i < 8; ++i) {
            const int  m    = rowBase + i + 8 * kh;
            const long gRow = (long)m * GRU_G;
            float xr = gx_t[gRow + col];
            float xz = gx_t[gRow + GRU_H + col];
            float xn = gx_t[gRow + 2 * GRU_H + col];
            float hp = hprev[(long)m * GRU_H + col];

            float r  = fsigmoid(xr + accR[i] + bhr);
            float z  = fsigmoid(xz + accZ[i] + bhz);
            float nn = tanhf(xn + r * (accN[i] + bhn));
            out_t[(long)m * GRU_H + col] = (1.0f - z) * nn + z * hp;
        }

        // ---- device-wide barrier between steps ----
        __threadfence();                 // make this block's out_t stores agent-visible
        __syncthreads();
        if (tid == 0) {
            __hip_atomic_fetch_add(cnt, 1u, __ATOMIC_RELEASE, __HIP_MEMORY_SCOPE_AGENT);
        }
        if (lane == 0) {                 // one spinner per wave: both CUs' L0s get the
            const unsigned target = (unsigned)(t + 1) * STEP_BLOCKS;   // acquire invalidate
            while (__hip_atomic_load(cnt, __ATOMIC_ACQUIRE, __HIP_MEMORY_SCOPE_AGENT) < target) {
                __builtin_amdgcn_s_sleep(2);
            }
        }
        __syncthreads();
    }
}

// ---------------------------------------------------------------------------
// Host side
// ---------------------------------------------------------------------------
extern "C" void kernel_launch(void* const* d_in, const int* in_sizes, int n_in,
                              void* d_out, int out_size, void* d_ws, size_t ws_size,
                              hipStream_t stream) {
    (void)in_sizes; (void)n_in; (void)ws_size;

    const float* x    = (const float*)d_in[0];   // [T][B][E]
    const float* hx   = (const float*)d_in[1];   // [B][H]
    const float* W_ih = (const float*)d_in[2];   // [G][E]
    const float* W_hh = (const float*)d_in[3];   // [G][H]
    const float* b_ih = (const float*)d_in[4];   // [G]
    const float* b_hh = (const float*)d_in[5];   // [G]
    float*       out  = (float*)d_out;           // [T][B][H]

    const int T = out_size / (GRU_B * GRU_H);

    // workspace layout (fixed offsets sized for T=512)
    char*  ws       = (char*)d_ws;
    size_t gxBytes  = (size_t)512 * GRU_B * GRU_G * sizeof(float);   // 384 MiB
    size_t xBytes   = (size_t)512 * GRU_B * GRU_E * sizeof(unsigned short);
    size_t wiBytes  = (size_t)GRU_G * GRU_E * sizeof(unsigned short);
    size_t whBytes  = (size_t)GRU_G * GRU_H * sizeof(unsigned short);

    float*          gx    = (float*)ws;
    unsigned short* xbf   = (unsigned short*)(ws + gxBytes);
    unsigned short* wihbf = (unsigned short*)(ws + gxBytes + xBytes);
    unsigned short* whhbf = (unsigned short*)(ws + gxBytes + xBytes + wiBytes);
    unsigned*       cnt   = (unsigned*)(ws + gxBytes + xBytes + wiBytes + whBytes);

    // 0) reset barrier counter (graph-capturable async memset)
    hipMemsetAsync(cnt, 0, sizeof(unsigned), stream);

    // 1) bf16 conversions
    cvt_f32_to_bf16<<<512, 256, 0, stream>>>(x, xbf, (long)T * GRU_B * GRU_E / 4);
    cvt_f32_to_bf16<<<96,  256, 0, stream>>>(W_ih, wihbf, (long)GRU_G * GRU_E / 4);
    cvt_f32_to_bf16<<<192, 256, 0, stream>>>(W_hh, whhbf, (long)GRU_G * GRU_H / 4);

    // 2) input-gate GEMM: gx = xbf @ wih^T + b_ih
    const int rowTiles = T * GRU_B / 16;
    gx_gemm<<<rowTiles * 24, 256, 0, stream>>>(xbf, wihbf, b_ih, gx);

    // 3) one persistent kernel runs the whole recurrence
    gru_persistent<<<STEP_BLOCKS, 128, 0, stream>>>(whhbf, hx, gx, b_hh, out, T, cnt);
}
